// DFNet011_49795850830313
// MI455X (gfx1250) — matrix operand order repaired and checked by reading, not compiled, not run on hardware
//
#include <hip/hip_runtime.h>
#include <hip/hip_bf16.h>

// ---------------------------------------------------------------------------
// DFNet dynamic-filter conv for gfx1250 (MI455X), bf16 WMMA implicit GEMM.
//   out[b] = (att0*Wc + att1*We) (*) x[b]  + (att0*bc + att1*be)
// Per-sample GEMM: M=256 (Cout), N=900 (30x30), K=2400 (96*5*5).
// ---------------------------------------------------------------------------

typedef __attribute__((ext_vector_type(16))) __bf16 v16bf;
typedef __attribute__((ext_vector_type(8)))  __bf16 bf16x8;
typedef __attribute__((ext_vector_type(8)))  float  v8f;
typedef __attribute__((ext_vector_type(4)))  float  f32x4;
typedef __attribute__((ext_vector_type(4)))  int    i32x4;

#define NB    128
#define CIN   96
#define HIN   63
#define WIN   63
#define COUT  256
#define HO    30
#define WO    30
#define NPIX  900      // HO*WO
#define KTOT  2400     // CIN*5*5
#define KC    32       // WMMA K per step
#define KITER 75       // KTOT/KC
#define LDK   40       // LDS row stride in bf16 (padded, keeps 16B align)

// ---------------------------------------------------------------------------
// Kernel 1: attention  (global avg pool -> fc1(24) relu -> fc2(2) -> softmax/T)
// ---------------------------------------------------------------------------
__global__ __launch_bounds__(256)
void dfnet_attention_kernel(const float* __restrict__ x,
                            const float* __restrict__ w1,   // [24,96]
                            const float* __restrict__ w2,   // [2,24]
                            float* __restrict__ att)        // [B,2]
{
    __shared__ float pooled[CIN];
    __shared__ float hidden[24];
    __shared__ float logits[2];

    const int b    = blockIdx.x;
    const int t    = threadIdx.x;
    const int wv   = t >> 5;
    const int lane = t & 31;
    const float* xb = x + (size_t)b * CIN * HIN * WIN;
    const float inv_hw = 1.0f / (float)(HIN * WIN);

    for (int ci = wv; ci < CIN; ci += 8) {
        const float* xc = xb + (size_t)ci * (HIN * WIN);
        float s = 0.0f;
        for (int p = lane; p < HIN * WIN; p += 32) s += xc[p];
        s += __shfl_xor(s, 16, 32);
        s += __shfl_xor(s,  8, 32);
        s += __shfl_xor(s,  4, 32);
        s += __shfl_xor(s,  2, 32);
        s += __shfl_xor(s,  1, 32);
        if (lane == 0) pooled[ci] = s * inv_hw;
    }
    __syncthreads();

    if (t < 24) {
        float h = 0.0f;
        #pragma unroll 8
        for (int ci = 0; ci < CIN; ++ci) h += pooled[ci] * w1[t * CIN + ci];
        hidden[t] = fmaxf(h, 0.0f);
    }
    __syncthreads();

    if (t < 2) {
        float l = 0.0f;
        #pragma unroll
        for (int j = 0; j < 24; ++j) l += hidden[j] * w2[t * 24 + j];
        logits[t] = l;
    }
    __syncthreads();

    if (t == 0) {
        const float invT = 1.0f / 31.0f;
        float l0 = logits[0] * invT, l1 = logits[1] * invT;
        float m  = fmaxf(l0, l1);
        float e0 = __expf(l0 - m), e1 = __expf(l1 - m);
        float r  = 1.0f / (e0 + e1);
        att[b * 2 + 0] = e0 * r;
        att[b * 2 + 1] = e1 * r;
    }
}

// ---------------------------------------------------------------------------
// Kernel 2: per-sample implicit GEMM with bf16 WMMA, f32 accumulate.
// grid = (ntile=8, mtile=2, b=128); block = 256 threads = 8 waves (4M x 2N)
// wave tile = 32(M) x 64(N) = 2x4 accumulators of 16x16.
// ---------------------------------------------------------------------------
__global__ __launch_bounds__(256, 2)
void dfnet_conv_wmma_kernel(const float* __restrict__ x,
                            const float* __restrict__ wc,   // [256,2400]
                            const float* __restrict__ we,   // [256,2400]
                            const float* __restrict__ cb,   // [256]
                            const float* __restrict__ eb,   // [256]
                            const float* __restrict__ att,  // [B,2]
                            float* __restrict__ out)        // [B,256,900]
{
    __shared__ __align__(16) __bf16 ldsA[2][128 * LDK];   // [Cout_local][K]
    __shared__ __align__(16) __bf16 ldsB[2][128 * LDK];   // [pix_local ][K]
    __shared__ __align__(16) int    offtab[KTOT];         // im2col k -> x offset

    const int b     = blockIdx.z;
    const int mtile = blockIdx.y;            // 0..1 (128 Cout each)
    const int ntile = blockIdx.x;            // 0..7 (128 pixels each)
    const int t     = threadIdx.x;

    const float att0 = att[b * 2 + 0];
    const float att1 = att[b * 2 + 1];
    const float* xb  = x + (size_t)b * CIN * HIN * WIN;

    // ---- one-time im2col offset table (pixel-independent part of address) ---
    for (int k = t; k < KTOT; k += 256) {
        const int ci = k / 25;
        const int r  = k % 25;
        offtab[k] = ci * (HIN * WIN) + (r / 5) * WIN + (r % 5);
    }

    // ---- cooperative loader mapping: 2 threads per row, 16 K each ----
    const int lrow = t >> 1;                 // 0..127
    const int lkb  = (t & 1) * 16;           // 0 or 16

    const int  pglob = ntile * 128 + lrow;   // pixel this thread feeds
    const bool pval  = (pglob < NPIX);
    // invalid rows read in-bounds garbage (column masked at store instead)
    const float* xp = pval ? (xb + (pglob / WO) * 2 * WIN + (pglob % WO) * 2) : xb;
    const float* wcp = wc + (size_t)(mtile * 128 + lrow) * KTOT + lkb;
    const float* wep = we + (size_t)(mtile * 128 + lrow) * KTOT + lkb;

    auto loadA = [&](int it, int s) {        // blend experts -> bf16 -> LDS
        const int k0 = it * KC;
        __bf16* dst = &ldsA[s][lrow * LDK + lkb];
        #pragma unroll
        for (int j = 0; j < 4; ++j) {
            f32x4 a = *(const f32x4*)(wcp + k0 + 4 * j);
            f32x4 e = *(const f32x4*)(wep + k0 + 4 * j);
            #pragma unroll
            for (int q = 0; q < 4; ++q)
                dst[4 * j + q] = (__bf16)(att0 * a[q] + att1 * e[q]);
        }
    };

    auto loadB = [&](int it, int s) {        // im2col gather -> bf16 -> LDS
        const int k0 = it * KC + lkb;
        __bf16* dst = &ldsB[s][lrow * LDK + lkb];
        i32x4 o[4];
        #pragma unroll
        for (int j = 0; j < 4; ++j)
            o[j] = *(const i32x4*)&offtab[k0 + 4 * j];
        #pragma unroll
        for (int j = 0; j < 4; ++j) {
            #pragma unroll
            for (int q = 0; q < 4; ++q)
                dst[4 * j + q] = (__bf16)xp[o[j][q]];
        }
    };

    // ---- wave / lane geometry ----
    const int wv   = t >> 5;
    const int lane = t & 31;
    const int wm   = wv & 3;                 // 4 M groups of 32 rows
    const int wn   = wv >> 2;                // 2 N groups of 64 cols
    const int lr   = lane & 15;
    const int hi   = lane >> 4;              // half-wave select

    v8f acc[2][4];
    #pragma unroll
    for (int mi = 0; mi < 2; ++mi)
        #pragma unroll
        for (int nj = 0; nj < 4; ++nj)
            acc[mi][nj] = (v8f){0.f,0.f,0.f,0.f,0.f,0.f,0.f,0.f};

    __syncthreads();                         // offtab ready
    loadA(0, 0);
    loadB(0, 0);
    __syncthreads();

    for (int it = 0; it < KITER; ++it) {
        const int s = it & 1;
        if (it + 1 < KITER) {                // prefetch next slice (double buffer)
            loadA(it + 1, s ^ 1);
            loadB(it + 1, s ^ 1);
        }
        if (it + 2 < KITER) {                // pull K+2 weight lines toward WGP
            __builtin_prefetch(wcp + (it + 2) * KC, 0, 3);
            __builtin_prefetch(wep + (it + 2) * KC, 0, 3);
        }

        // A fragments: lanes 0-15 take K {0..7,16..23}, lanes 16-31 K {8..15,24..31}
        v16bf afr[2];
        #pragma unroll
        for (int mi = 0; mi < 2; ++mi) {
            const int m = wm * 32 + mi * 16 + lr;
            const __bf16* base = &ldsA[s][m * LDK + hi * 8];
            bf16x8 lo  = *(const bf16x8*)(base);
            bf16x8 hi8 = *(const bf16x8*)(base + 16);
            afr[mi] = __builtin_shufflevector(lo, hi8,
                0,1,2,3,4,5,6,7,8,9,10,11,12,13,14,15);
        }
        // B fragments: column n per lane, lanes 0-15 take K {0..7,16..23}, lanes 16-31 K {8..15,24..31}
        v16bf bfr[4];
        #pragma unroll
        for (int nj = 0; nj < 4; ++nj) {
            const int n = wn * 64 + nj * 16 + lr;
            const __bf16* base = &ldsB[s][n * LDK + hi * 8];
            bf16x8 lo  = *(const bf16x8*)(base);
            bf16x8 hi8 = *(const bf16x8*)(base + 16);
            bfr[nj] = __builtin_shufflevector(lo, hi8,
                0,1,2,3,4,5,6,7,8,9,10,11,12,13,14,15);
        }

        #pragma unroll
        for (int mi = 0; mi < 2; ++mi)
            #pragma unroll
            for (int nj = 0; nj < 4; ++nj)
                acc[mi][nj] = __builtin_amdgcn_wmma_f32_16x16x32_bf16(
                    false, afr[mi], false, bfr[nj],
                    (short)0, acc[mi][nj], false, false);

        __syncthreads();
    }

    // ---- epilogue: blended bias + store (C layout: lane<16 -> M=r, else M=r+8)
    #pragma unroll
    for (int mi = 0; mi < 2; ++mi) {
        #pragma unroll
        for (int r = 0; r < 8; ++r) {
            const int co = mtile * 128 + wm * 32 + mi * 16 + hi * 8 + r;
            const float bias = att0 * cb[co] + att1 * eb[co];
            float* orow = out + ((size_t)b * COUT + co) * NPIX;
            #pragma unroll
            for (int nj = 0; nj < 4; ++nj) {
                const int n = ntile * 128 + wn * 64 + nj * 16 + lr;
                if (n < NPIX) orow[n] = acc[mi][nj][r] + bias;
            }
        }
    }
}

// ---------------------------------------------------------------------------
extern "C" void kernel_launch(void* const* d_in, const int* in_sizes, int n_in,
                              void* d_out, int out_size, void* d_ws, size_t ws_size,
                              hipStream_t stream) {
    const float* x  = (const float*)d_in[0];   // [128,96,63,63]
    const float* wc = (const float*)d_in[1];   // [1,256,96,5,5]
    const float* cb = (const float*)d_in[2];   // [1,256]
    const float* we = (const float*)d_in[3];   // [1,256,96,5,5]
    const float* eb = (const float*)d_in[4];   // [1,256]
    const float* w1 = (const float*)d_in[5];   // [24,96]
    const float* w2 = (const float*)d_in[6];   // [2,24]
    float* out = (float*)d_out;                // [128,256,30,30]
    float* att = (float*)d_ws;                 // [128,2] scratch

    dfnet_attention_kernel<<<NB, 256, 0, stream>>>(x, w1, w2, att);

    dim3 grid(8 /*ntiles*/, 2 /*mtiles*/, NB);
    dfnet_conv_wmma_kernel<<<grid, 256, 0, stream>>>(x, wc, we, cb, eb, att, out);
}
